// SparseBlockSeries_25211458027793
// MI455X (gfx1250) — matrix operand order repaired and checked by reading, not compiled, not run on hardware
//
#include <hip/hip_runtime.h>

typedef __attribute__((ext_vector_type(16))) _Float16     v16h;
typedef __attribute__((ext_vector_type(8)))  float        v8f;
typedef __attribute__((ext_vector_type(4)))  unsigned int v4u;
typedef __attribute__((ext_vector_type(8)))  unsigned int v8u;

#define NCH 64
#define KNB 9
#define LEAK 0.33f
#define EPSV 1e-5f
#define MTILES 4   // 4 x 16 = 64 rows per wave -> each B fragment feeds 4 WMMAs

// ---------------- utility kernels ----------------

__global__ void zero_f32(float* __restrict__ p, int n) {
    int i = blockIdx.x * blockDim.x + threadIdx.x;
    if (i < n) p[i] = 0.f;
}

__global__ void cvt_x16(const float* __restrict__ in, _Float16* __restrict__ o, int total) {
    int i = blockIdx.x * blockDim.x + threadIdx.x;
    if (i < total) o[i] = (_Float16)in[i];
}

// Convert both conv weight tensors [4][9][64][64] (f32, [b][k][cin][cout]) into
// f16, transposed per conv to [k][d][cin] so WMMA B-fragments are 16 contiguous
// halves (one 32B load per fragment). conv id: 2b = conv1 of block b, 2b+1 = conv2.
__global__ void cvt_weights(const float* __restrict__ w1, const float* __restrict__ w2,
                            _Float16* __restrict__ w16t, int total) {
    int e = blockIdx.x * blockDim.x + threadIdx.x;
    if (e >= total) return;
    int c    = e & 63;          // input channel (GEMM K dim)
    int d    = (e >> 6) & 63;   // output channel
    int k    = (e >> 12) % KNB; // neighbor
    int conv = e / (KNB * 64 * 64);
    int b    = conv >> 1;
    const float* src = (conv & 1) ? w2 : w1;
    w16t[e] = (_Float16)src[(((size_t)b * KNB + k) * 64 + c) * 64 + d];
}

// ---------------- WMMA submanifold conv ----------------
// One wave owns a 64-row x 64-channel output chunk: 4 M-tiles x 4 N-tiles of
// 16x16 f32 accumulators. Weights (B) are pre-converted f16 loaded once per
// chunk and reused across the 4 M-tiles. The input tensor is consumed from a
// pre-materialized f16 copy, so gathers are direct B128 loads of halves with
// the neighbor mask applied as a bitwise 0/~0 select (no converts in the hot
// loop, EXEC stays all-1 for WMMA). Per-channel sum / sum-of-squares for the
// following BatchNorm are accumulated in registers and atomically merged.
__global__ __launch_bounds__(256)
void subm_conv_wmma(const _Float16* __restrict__ x16,   // [N][64] f16 activations
                    const int* __restrict__ nidx,
                    const unsigned char* __restrict__ nmask,
                    const _Float16* __restrict__ w16t,   // [9][64][64] f16, [k][d][c]
                    float* __restrict__ out,             // [N][64] f32 (pre-BN)
                    float* __restrict__ stats,           // [128]: sum[64], sumsq[64]
                    int nchunks) {
    const int lane = threadIdx.x & 31;
    const int half = lane >> 4;   // 0: K-halves {0..15}, 1: {16..31} of each 32-ch chunk
    const int mrow = lane & 15;
    const int wavesPerBlk = blockDim.x >> 5;
    const int gwave  = blockIdx.x * wavesPerBlk + (threadIdx.x >> 5);
    const int nwaves = gridDim.x * wavesPerBlk;

    float ssum[4] = {0.f, 0.f, 0.f, 0.f};
    float ssq[4]  = {0.f, 0.f, 0.f, 0.f};

    for (int chunk = gwave; chunk < nchunks; chunk += nwaves) {
        const int n0 = chunk * (16 * MTILES);
        v8f acc[MTILES][4] = {};

        #pragma unroll
        for (int k = 0; k < KNB; ++k) {
            // Gather A fragments (f16 rows) for all 4 M-tiles of this neighbor.
            v16h a[MTILES][2];
            #pragma unroll
            for (int mt = 0; mt < MTILES; ++mt) {
                const int row      = n0 + mt * 16 + mrow;
                const int nb       = nidx[(size_t)row * KNB + k];
                const unsigned sel = nmask[(size_t)row * KNB + k] ? 0xFFFFFFFFu : 0u;
                const _Float16* src = x16 + (size_t)nb * NCH + half * 16;
                // half-groups (rel. to half*16): {0..7},{8..15},{32..39},{40..47}
                v4u g[4];
                #pragma unroll
                for (int gi = 0; gi < 4; ++gi)
                    g[gi] = *(const v4u*)(src + 8 * gi + 16 * (gi >> 1)) & sel;
                #pragma unroll
                for (int kc = 0; kc < 2; ++kc) {
                    v8u r;
                    #pragma unroll
                    for (int w = 0; w < 4; ++w) {
                        r[w]     = g[2 * kc][w];
                        r[4 + w] = g[2 * kc + 1][w];
                    }
                    a[mt][kc] = __builtin_bit_cast(v16h, r);
                }
            }

            // One B fragment per (kc, nt), reused across the 4 M-tiles.
            #pragma unroll
            for (int kc = 0; kc < 2; ++kc) {
                const _Float16* wk = w16t + (size_t)k * 64 * 64 + kc * 32 + half * 16;
                #pragma unroll
                for (int nt = 0; nt < 4; ++nt) {
                    const int d = nt * 16 + mrow;
                    v16h bfrag = *(const v16h*)(wk + d * 64);
                    #pragma unroll
                    for (int mt = 0; mt < MTILES; ++mt)
                        acc[mt][nt] = __builtin_amdgcn_wmma_f32_16x16x32_f16(
                            false, a[mt][kc], false, bfrag, (short)0, acc[mt][nt],
                            false, false);
                }
            }
        }

        // C/D layout: lane holds channel d = nt*16 + mrow, rows j + half*8.
        #pragma unroll
        for (int mt = 0; mt < MTILES; ++mt) {
            #pragma unroll
            for (int nt = 0; nt < 4; ++nt) {
                const int d = nt * 16 + mrow;
                float s = 0.f, q = 0.f;
                #pragma unroll
                for (int j = 0; j < 8; ++j) {
                    float v = acc[mt][nt][j];
                    out[(size_t)(n0 + mt * 16 + half * 8 + j) * NCH + d] = v;
                    s += v; q += v * v;
                }
                ssum[nt] += s; ssq[nt] += q;
            }
        }
    }

    // Lane pair (l, l+16) shares channel d: combine, then one atomic per channel.
    #pragma unroll
    for (int nt = 0; nt < 4; ++nt) {
        float s = ssum[nt] + __shfl_xor(ssum[nt], 16);
        float q = ssq[nt]  + __shfl_xor(ssq[nt], 16);
        if (half == 0) {
            atomicAdd(&stats[nt * 16 + mrow], s);
            atomicAdd(&stats[64 + nt * 16 + mrow], q);
        }
    }
}

// ---------------- BatchNorm helpers ----------------

__global__ void bn_finalize(const float* __restrict__ stats,
                            const float* __restrict__ gamma,
                            const float* __restrict__ beta,
                            float* __restrict__ sb, float invN) {
    int d = threadIdx.x;
    if (d < NCH) {
        float mean = stats[d] * invN;
        float var  = stats[64 + d] * invN - mean * mean;
        float sc   = gamma[d] * rsqrtf(var + EPSV);
        sb[d]      = sc;
        sb[64 + d] = beta[d] - mean * sc;
    }
}

// BN + LeakyReLU; emits the f16 tensor consumed by the next conv's gathers.
__global__ void bn_leaky_f16(const float* __restrict__ t, const float* __restrict__ sb,
                             _Float16* __restrict__ o16, int total) {
    int i = blockIdx.x * blockDim.x + threadIdx.x;
    if (i >= total) return;
    int d = i & 63;
    float v = t[i] * sb[d] + sb[64 + d];
    v = v > 0.f ? v : LEAK * v;
    o16[i] = (_Float16)v;
}

// BN + residual + LeakyReLU; writes the fp32 master copy (residual chain /
// final output) and the f16 copy for the next conv's gathers.
__global__ void bn_res_leaky(const float* __restrict__ t, const float* __restrict__ sb,
                             const float* __restrict__ res,
                             float* __restrict__ xo, _Float16* __restrict__ xo16,
                             int total) {
    int i = blockIdx.x * blockDim.x + threadIdx.x;
    if (i >= total) return;
    int d = i & 63;
    float v = t[i] * sb[d] + sb[64 + d] + res[i];
    v = v > 0.f ? v : LEAK * v;
    xo[i]   = v;
    xo16[i] = (_Float16)v;
}

// ---------------- driver ----------------

extern "C" void kernel_launch(void* const* d_in, const int* in_sizes, int n_in,
                              void* d_out, int out_size, void* d_ws, size_t ws_size,
                              hipStream_t stream) {
    const float*         features = (const float*)d_in[0];
    const int*           nidx     = (const int*)d_in[1];
    const unsigned char* nmask    = (const unsigned char*)d_in[2]; // jax bool = 1 byte
    const float* w1  = (const float*)d_in[3];
    const float* g1  = (const float*)d_in[4];
    const float* bt1 = (const float*)d_in[5];
    const float* w2  = (const float*)d_in[6];
    const float* g2  = (const float*)d_in[7];
    const float* bt2 = (const float*)d_in[8];

    const int N       = in_sizes[0] / NCH;   // 131072
    const int nchunks = N / (16 * MTILES);   // 2048 64-row chunks
    const int total   = N * NCH;
    const float invN  = 1.f / (float)N;

    // workspace layout
    char* ws = (char*)d_ws;
    _Float16* w16t = (_Float16*)ws;                              // 8*9*64*64 halves
    size_t off = ((8ull * KNB * 64 * 64 * sizeof(_Float16)) + 255) & ~255ull;
    float* stats = (float*)(ws + off); off += 8 * 128 * sizeof(float);
    float* sb    = (float*)(ws + off); off += 8 * 128 * sizeof(float);
    off = (off + 255) & ~255ull;
    float* tmp1 = (float*)(ws + off); off += (size_t)total * sizeof(float);
    float* tmp2 = (float*)(ws + off); off += (size_t)total * sizeof(float);
    float* xbuf = (float*)(ws + off); off += (size_t)total * sizeof(float);
    _Float16* feat16 = (_Float16*)(ws + off); off += (size_t)total * sizeof(_Float16);
    _Float16* t16    = (_Float16*)(ws + off); off += (size_t)total * sizeof(_Float16);
    _Float16* x16    = (_Float16*)(ws + off); off += (size_t)total * sizeof(_Float16);

    const int egrid = (total + 255) / 256;

    // stats accumulate via atomics -> must be zeroed every call (graph replays)
    zero_f32<<<(8 * 128 + 255) / 256, 256, 0, stream>>>(stats, 8 * 128);
    const int wtot = 8 * KNB * 64 * 64;
    cvt_weights<<<(wtot + 255) / 256, 256, 0, stream>>>(w1, w2, w16t, wtot);
    cvt_x16<<<egrid, 256, 0, stream>>>(features, feat16, total);

    const dim3 cgrid(256), cblk(256);           // 2048 waves <-> 2048 chunks

    const float*    curx   = features;  // fp32 residual chain
    const _Float16* curx16 = feat16;    // f16 tensor consumed by conv gathers
    for (int b = 0; b < 4; ++b) {
        const int c1 = 2 * b, c2 = 2 * b + 1;
        subm_conv_wmma<<<cgrid, cblk, 0, stream>>>(
            curx16, nidx, nmask, w16t + (size_t)c1 * KNB * 64 * 64, tmp1, stats + c1 * 128, nchunks);
        bn_finalize<<<1, 64, 0, stream>>>(stats + c1 * 128, g1 + b * 64, bt1 + b * 64,
                                          sb + c1 * 128, invN);
        bn_leaky_f16<<<egrid, 256, 0, stream>>>(tmp1, sb + c1 * 128, t16, total);

        subm_conv_wmma<<<cgrid, cblk, 0, stream>>>(
            t16, nidx, nmask, w16t + (size_t)c2 * KNB * 64 * 64, tmp2, stats + c2 * 128, nchunks);
        bn_finalize<<<1, 64, 0, stream>>>(stats + c2 * 128, g2 + b * 64, bt2 + b * 64,
                                          sb + c2 * 128, invN);

        float* xo = (b == 3) ? (float*)d_out : xbuf;
        bn_res_leaky<<<egrid, 256, 0, stream>>>(tmp2, sb + c2 * 128, curx, xo, x16, total);
        curx   = xo;
        curx16 = x16;
    }
}
